// UnifiedMultiHeadAttention_52020643889184
// MI455X (gfx1250) — compile-verified, matrix operand order checked by reading, not run on hardware
//
#include <hip/hip_runtime.h>
#include <hip/hip_bf16.h>

typedef __bf16 bf16_t;
typedef __attribute__((ext_vector_type(16))) __bf16 v16bf;
typedef __attribute__((ext_vector_type(8)))  __bf16 v8bf;
typedef __attribute__((ext_vector_type(8)))  float  v8f;
typedef __attribute__((ext_vector_type(4)))  float  v4f;
typedef __attribute__((ext_vector_type(2)))  unsigned int v2u;
typedef __attribute__((ext_vector_type(4)))  unsigned int v4u;

#define DIM   512
#define NH    8
#define HD    64
#define NTOK  1024
#define BATCH 8

__device__ __forceinline__ v8f wmma_bf16(v16bf a, v16bf b, v8f c) {
  return __builtin_amdgcn_wmma_f32_16x16x32_bf16(false, a, false, b, (short)0, c,
                                                 false, false);
}

// Unified fragment load: 16 rows x 32 contiguous-K bf16, leading dim ld.
// ld must be a multiple of 8 and base 16B-aligned -> two b128 loads per lane.
__device__ __forceinline__ v16bf load_frag(const bf16_t* base, int ld) {
  const int lane = threadIdx.x & 31;
  const int m    = lane & 15;
  const int koff = (lane >> 4) << 3;
  const bf16_t* p = base + m * ld + koff;
  v8bf lo = *(const v8bf*)p;
  v8bf hi = *(const v8bf*)(p + 16);
  return __builtin_shufflevector(lo, hi, 0, 1, 2, 3, 4, 5, 6, 7,
                                 8, 9, 10, 11, 12, 13, 14, 15);
}

__device__ __forceinline__ unsigned int pack_bf16(float a, float b) {
  union { bf16_t h[2]; unsigned int u; } p;
  p.h[0] = (bf16_t)a; p.h[1] = (bf16_t)b;
  return p.u;
}

// ---------------------------------------------------------------- phase 1
// Fused QKV projection per (b,h): X[128x512]*W[512x64] x3, bias, Q scaled.
// Q,K stored bf16 [bh][n][d]; V stored transposed [bh][d][n].
#define LDX  40   // xs  [n][c]
#define LDWT 40   // w^T [d][c]
#define LDS1 72   // per-wave store-stage [m][d]

__global__ __launch_bounds__(256) void qkv_proj_kernel(
    const float* __restrict__ query,
    const float* __restrict__ Wq, const float* __restrict__ bq,
    const float* __restrict__ Wk, const float* __restrict__ bk,
    const float* __restrict__ Wv, const float* __restrict__ bv,
    bf16_t* __restrict__ Qb, bf16_t* __restrict__ Kb, bf16_t* __restrict__ Vt) {
  __shared__ __align__(16) bf16_t xs[128 * LDX];
  __shared__ __align__(16) bf16_t wqs[HD * LDWT];
  __shared__ __align__(16) bf16_t wks[HD * LDWT];
  __shared__ __align__(16) bf16_t wvs[HD * LDWT];
  __shared__ __align__(16) bf16_t stg[8 * 16 * LDS1];

  const int t  = threadIdx.x;
  const int n0 = blockIdx.x * 128;
  const int bh = blockIdx.y;
  const int b  = bh / NH, h = bh % NH;

  const float* xbase = query + (size_t)b * DIM * NTOK;  // [C][N]
  const float* wqb   = Wq + (size_t)h * DIM * HD;
  const float* wkb   = Wk + (size_t)h * DIM * HD;
  const float* wvb   = Wv + (size_t)h * DIM * HD;

  const int w    = t >> 5;
  const int lane = t & 31;
  const int m0   = w * 16;

  v8f aq[4], ak[4], av[4];
#pragma unroll
  for (int j = 0; j < 4; ++j) { aq[j] = {}; ak[j] = {}; av[j] = {}; }

  for (int kk = 0; kk < DIM / 32; ++kk) {
    const int c0 = kk * 32;
    __syncthreads();
    // X tile 128n x 32c, coalesced along n, pair-packed b32 LDS stores
#pragma unroll
    for (int i = 0; i < 8; ++i) {
      int idx = t + i * 256;                 // 2048 pairs
      int nn = idx & 127, cc = (idx >> 7) * 2;
      float f0 = xbase[(size_t)(c0 + cc) * NTOK + n0 + nn];
      float f1 = xbase[(size_t)(c0 + cc + 1) * NTOK + n0 + nn];
      *(unsigned int*)(xs + nn * LDX + cc) = pack_bf16(f0, f1);
    }
    // Weight tiles, transposed to [d][c], pair-packed over c
#pragma unroll
    for (int i = 0; i < 4; ++i) {
      int idx = t + i * 256;                 // 1024 pairs
      int dd = idx & 63, cc = (idx >> 6) * 2;
      size_t g0 = (size_t)(c0 + cc) * HD + dd, g1 = g0 + HD;
      *(unsigned int*)(wqs + dd * LDWT + cc) = pack_bf16(wqb[g0], wqb[g1]);
      *(unsigned int*)(wks + dd * LDWT + cc) = pack_bf16(wkb[g0], wkb[g1]);
      *(unsigned int*)(wvs + dd * LDWT + cc) = pack_bf16(wvb[g0], wvb[g1]);
    }
    __syncthreads();

    v16bf a = load_frag(xs + m0 * LDX, LDX);
#pragma unroll
    for (int j = 0; j < 4; ++j) {
      aq[j] = wmma_bf16(a, load_frag(wqs + (j * 16) * LDWT, LDWT), aq[j]);
      ak[j] = wmma_bf16(a, load_frag(wks + (j * 16) * LDWT, LDWT), ak[j]);
      av[j] = wmma_bf16(a, load_frag(wvs + (j * 16) * LDWT, LDWT), av[j]);
    }
  }

  const float scale = 0.125f;  // HEAD_DIM^-0.5 folded into Q
  const int col  = lane & 15;
  const int drow = (lane >> 4) * 8;
  bf16_t* sw = stg + w * 16 * LDS1;

  // ---- Q: stage [m][d] per-wave, then coalesced b128 stores
#pragma unroll
  for (int j = 0; j < 4; ++j) {
    const float bqv = bq[h * HD + j * 16 + col];
#pragma unroll
    for (int r = 0; r < 8; ++r)
      sw[(drow + r) * LDS1 + j * 16 + col] = (bf16_t)((aq[j][r] + bqv) * scale);
  }
  {
    bf16_t* qout = Qb + ((size_t)bh * NTOK + n0 + m0) * HD;
#pragma unroll
    for (int q = 0; q < 4; ++q) {
      int chunk = lane + q * 32;
      int mloc = chunk >> 3, c8 = (chunk & 7) * 8;
      *(v8bf*)(qout + (size_t)mloc * HD + c8) = *(const v8bf*)(sw + mloc * LDS1 + c8);
    }
  }
  // ---- K: same staging path
#pragma unroll
  for (int j = 0; j < 4; ++j) {
    const float bkv = bk[h * HD + j * 16 + col];
#pragma unroll
    for (int r = 0; r < 8; ++r)
      sw[(drow + r) * LDS1 + j * 16 + col] = (bf16_t)(ak[j][r] + bkv);
  }
  {
    bf16_t* kout = Kb + ((size_t)bh * NTOK + n0 + m0) * HD;
#pragma unroll
    for (int q = 0; q < 4; ++q) {
      int chunk = lane + q * 32;
      int mloc = chunk >> 3, c8 = (chunk & 7) * 8;
      *(v8bf*)(kout + (size_t)mloc * HD + c8) = *(const v8bf*)(sw + mloc * LDS1 + c8);
    }
  }
  // ---- V: direct transposed [d][n] b128 stores (8 rows contiguous per lane)
  {
    bf16_t* vtout = Vt + (size_t)bh * HD * NTOK;
#pragma unroll
    for (int j = 0; j < 4; ++j) {
      const int d = j * 16 + col;
      const float bvv = bv[h * HD + d];
      v8bf pv;
#pragma unroll
      for (int r = 0; r < 8; ++r) pv[r] = (bf16_t)(av[j][r] + bvv);
      *(v8bf*)(vtout + (size_t)d * NTOK + n0 + m0 + drow) = pv;
    }
  }
}

// ---------------------------------------------------------------- phase 2
// Flash attention per (b,h): 128 q-rows/block, 64-token KV chunks,
// online softmax, O = P*V.
#define LKV2 72   // Ks  [token][d]   (144B rows, 16B aligned)
#define LDV2 72   // Vts [d][token]   (144B rows, 16B aligned)
#define LDP2 72   // per-wave stage: P [m][k64] and output [m][d]

__global__ __launch_bounds__(256) void attention_kernel(
    const bf16_t* __restrict__ Qb, const bf16_t* __restrict__ Kb,
    const bf16_t* __restrict__ Vt, bf16_t* __restrict__ xo) {
  __shared__ __align__(16) bf16_t Ks[64 * LKV2];
  __shared__ __align__(16) bf16_t Vts[HD * LDV2];
  __shared__ __align__(16) bf16_t Pst[8 * 16 * LDP2];

  const int t    = threadIdx.x;
  const int bh   = blockIdx.y;
  const int b    = bh / NH, h = bh % NH;
  const int n0   = blockIdx.x * 128;
  const int w    = t >> 5;
  const int lane = t & 31;
  const int m0   = n0 + w * 16;

  // Q fragments in registers for the whole pass (b128 global loads)
  const bf16_t* qptr = Qb + ((size_t)bh * NTOK + m0) * HD;
  const v16bf qa0 = load_frag(qptr, HD);
  const v16bf qa1 = load_frag(qptr + 32, HD);

  v8f o0 = {}, o1 = {}, o2 = {}, o3 = {};
  float rmax[8], rsum[8];
#pragma unroll
  for (int r = 0; r < 8; ++r) { rmax[r] = -3.0e38f; rsum[r] = 0.0f; }

  const bf16_t* kbase  = Kb + (size_t)bh * NTOK * HD;   // [token][d]
  const bf16_t* vtbase = Vt + (size_t)bh * HD * NTOK;   // [d][token]
  bf16_t* pw = Pst + w * 16 * LDP2;
  const int col  = lane & 15;
  const int drow = (lane >> 4) * 8;

  for (int kc = 0; kc < NTOK / 64; ++kc) {
    __syncthreads();
    // K chunk: 64 tokens x 64d, b128 copies (512 x 16B)
#pragma unroll
    for (int i = 0; i < 2; ++i) {
      int chunk = t + i * 256;
      int kk = chunk >> 3, c4 = (chunk & 7) * 8;
      *(v4u*)(Ks + kk * LKV2 + c4) =
          *(const v4u*)(kbase + (size_t)(kc * 64 + kk) * HD + c4);
    }
    // V chunk (transposed): 64d x 64 tokens, b128 copies (512 x 16B)
#pragma unroll
    for (int i = 0; i < 2; ++i) {
      int chunk = t + i * 256;
      int dd = chunk >> 3, c4 = (chunk & 7) * 8;
      *(v4u*)(Vts + dd * LDV2 + c4) =
          *(const v4u*)(vtbase + (size_t)dd * NTOK + kc * 64 + c4);
    }
    if (kc + 1 < NTOK / 64) {
      __builtin_prefetch(kbase + (size_t)(kc + 1) * 64 * HD + t * 32, 0, 0);
      __builtin_prefetch(vtbase + (size_t)(kc + 1) * 64 + (t & 63) * NTOK, 0, 0);
    }
    __syncthreads();

    // S = Q * K^T : four 16x16 score tiles covering 64 tokens
    v8f s[4];
#pragma unroll
    for (int tt = 0; tt < 4; ++tt) {
      const bf16_t* kb = Ks + (tt * 16) * LKV2;
      v8f st = {};
      st = wmma_bf16(qa0, load_frag(kb, LKV2), st);
      st = wmma_bf16(qa1, load_frag(kb + 32, LKV2), st);
      s[tt] = st;
    }

    // Online softmax across 16-lane column groups
#pragma unroll
    for (int r = 0; r < 8; ++r) {
      float mx = fmaxf(fmaxf(s[0][r], s[1][r]), fmaxf(s[2][r], s[3][r]));
#pragma unroll
      for (int off = 1; off < 16; off <<= 1) mx = fmaxf(mx, __shfl_xor(mx, off));
      float nm   = fmaxf(rmax[r], mx);
      float corr = __expf(rmax[r] - nm);
      rmax[r] = nm;
      float p0 = __expf(s[0][r] - nm);
      float p1 = __expf(s[1][r] - nm);
      float p2 = __expf(s[2][r] - nm);
      float p3 = __expf(s[3][r] - nm);
      s[0][r] = p0; s[1][r] = p1; s[2][r] = p2; s[3][r] = p3;
      float ps = (p0 + p1) + (p2 + p3);
#pragma unroll
      for (int off = 1; off < 16; off <<= 1) ps += __shfl_xor(ps, off);
      rsum[r] = rsum[r] * corr + ps;
      o0[r] = o0[r] * corr; o1[r] = o1[r] * corr;
      o2[r] = o2[r] * corr; o3[r] = o3[r] * corr;
    }

    // P: C-layout -> A-layout via per-wave LDS, then O += P*V (k = 64 tokens)
#pragma unroll
    for (int r = 0; r < 8; ++r) {
      bf16_t* prow = pw + (drow + r) * LDP2;
      prow[0  + col] = (bf16_t)s[0][r];
      prow[16 + col] = (bf16_t)s[1][r];
      prow[32 + col] = (bf16_t)s[2][r];
      prow[48 + col] = (bf16_t)s[3][r];
    }
    v16bf ap0 = load_frag(pw, LDP2);
    v16bf ap1 = load_frag(pw + 32, LDP2);
    o0 = wmma_bf16(ap0, load_frag(Vts + 0 * LDV2,       LDV2), o0);
    o0 = wmma_bf16(ap1, load_frag(Vts + 0 * LDV2 + 32,  LDV2), o0);
    o1 = wmma_bf16(ap0, load_frag(Vts + 16 * LDV2,      LDV2), o1);
    o1 = wmma_bf16(ap1, load_frag(Vts + 16 * LDV2 + 32, LDV2), o1);
    o2 = wmma_bf16(ap0, load_frag(Vts + 32 * LDV2,      LDV2), o2);
    o2 = wmma_bf16(ap1, load_frag(Vts + 32 * LDV2 + 32, LDV2), o2);
    o3 = wmma_bf16(ap0, load_frag(Vts + 48 * LDV2,      LDV2), o3);
    o3 = wmma_bf16(ap1, load_frag(Vts + 48 * LDV2 + 32, LDV2), o3);
  }

  // Epilogue: normalize, stage [m][d] per-wave, coalesced b128 stores
  float inv[8];
#pragma unroll
  for (int r = 0; r < 8; ++r) inv[r] = 1.0f / rsum[r];
#pragma unroll
  for (int r = 0; r < 8; ++r) {
    bf16_t* row = pw + (drow + r) * LDP2;
    row[0  + col] = (bf16_t)(o0[r] * inv[r]);
    row[16 + col] = (bf16_t)(o1[r] * inv[r]);
    row[32 + col] = (bf16_t)(o2[r] * inv[r]);
    row[48 + col] = (bf16_t)(o3[r] * inv[r]);
  }
  bf16_t* obase = xo + ((size_t)b * NTOK + m0) * DIM + h * HD;
#pragma unroll
  for (int q = 0; q < 4; ++q) {
    int chunk = lane + q * 32;
    int mloc = chunk >> 3, c8 = (chunk & 7) * 8;
    *(v8bf*)(obase + (size_t)mloc * DIM + c8) = *(const v8bf*)(pw + mloc * LDP2 + c8);
  }
}

// ---------------------------------------------------------------- phase 3
// out = xo[8192x512] @ Wo[512x512] + bo -> [B][C][N] via LDS transpose tile.
#define LDA3 40    // as  [n][k]
#define LDW3 40    // Wo^T[c][k]
#define LDO3 132   // os  [c][n] f32

__global__ __launch_bounds__(256) void out_proj_kernel(
    const bf16_t* __restrict__ xo, const float* __restrict__ Wo,
    const float* __restrict__ bo, float* __restrict__ out) {
  __shared__ __align__(16) bf16_t as[128 * LDA3];
  __shared__ __align__(16) bf16_t wot[HD * LDW3];
  __shared__ __align__(16) float  os[64 * LDO3];

  const int t    = threadIdx.x;
  const int mb   = blockIdx.x * 128;
  const int c0   = blockIdx.y * 64;
  const int w    = t >> 5;
  const int lane = t & 31;
  const int m0   = w * 16;

  v8f acc[4];
#pragma unroll
  for (int j = 0; j < 4; ++j) acc[j] = {};

  for (int kk = 0; kk < DIM / 32; ++kk) {
    const int k0 = kk * 32;
    __syncthreads();
#pragma unroll
    for (int i = 0; i < 4; ++i) {
      int chunk = t + i * 256;                    // 1024 x 8B
      int nn = chunk >> 3, c2 = (chunk & 7) * 4;
      *(v2u*)(as + nn * LDA3 + c2) =
          *(const v2u*)(xo + (size_t)(mb + nn) * DIM + k0 + c2);
    }
#pragma unroll
    for (int i = 0; i < 4; ++i) {
      int idx = t + i * 256;                      // 1024 pairs
      int dd = idx & 63, cc = (idx >> 6) * 2;
      *(unsigned int*)(wot + dd * LDW3 + cc) =
          pack_bf16(Wo[(size_t)(k0 + cc) * DIM + c0 + dd],
                    Wo[(size_t)(k0 + cc + 1) * DIM + c0 + dd]);
    }
    __syncthreads();
    v16bf a = load_frag(as + m0 * LDA3, LDA3);
#pragma unroll
    for (int j = 0; j < 4; ++j)
      acc[j] = wmma_bf16(a, load_frag(wot + (j * 16) * LDW3, LDW3), acc[j]);
  }

  __syncthreads();
  const int col  = lane & 15;
  const int drow = (lane >> 4) * 8;
#pragma unroll
  for (int j = 0; j < 4; ++j) {
    float* row = os + (j * 16 + col) * LDO3 + m0 + drow;
    v4f lo = {acc[j][0], acc[j][1], acc[j][2], acc[j][3]};
    v4f hi = {acc[j][4], acc[j][5], acc[j][6], acc[j][7]};
    *(v4f*)(row)     = lo;
    *(v4f*)(row + 4) = hi;
  }
  __syncthreads();

  const int bidx  = mb >> 10;
  const int nbase = mb & 1023;
#pragma unroll
  for (int i = 0; i < 8; ++i) {
    int chunk = t + i * 256;                      // 2048 x 16B
    int cl = chunk >> 5, c4 = (chunk & 31) * 4;
    v4f vv = *(const v4f*)(os + cl * LDO3 + c4);
    vv += bo[c0 + cl];
    *(v4f*)(out + ((size_t)bidx * DIM + c0 + cl) * NTOK + nbase + c4) = vv;
  }
}

// ---------------------------------------------------------------- launch
extern "C" void kernel_launch(void* const* d_in, const int* in_sizes, int n_in,
                              void* d_out, int out_size, void* d_ws, size_t ws_size,
                              hipStream_t stream) {
  (void)in_sizes; (void)n_in; (void)out_size; (void)ws_size;
  const float* query = (const float*)d_in[0];
  const float* Wq = (const float*)d_in[1];
  const float* bq = (const float*)d_in[2];
  const float* Wk = (const float*)d_in[3];
  const float* bk = (const float*)d_in[4];
  const float* Wv = (const float*)d_in[5];
  const float* bv = (const float*)d_in[6];
  const float* Wo = (const float*)d_in[7];
  const float* bo = (const float*)d_in[8];
  float* out = (float*)d_out;

  const size_t qkv_elems = (size_t)BATCH * NH * NTOK * HD;  // 4M bf16 each
  bf16_t* Qb = (bf16_t*)d_ws;
  bf16_t* Kb = Qb + qkv_elems;
  bf16_t* Vt = Kb + qkv_elems;                    // [bh][d][n] transposed
  bf16_t* xo = Vt + qkv_elems;                    // [B*N][DIM] concat heads

  dim3 blk(256);
  qkv_proj_kernel<<<dim3(NTOK / 128, BATCH * NH), blk, 0, stream>>>(
      query, Wq, bq, Wk, bk, Wv, bv, Qb, Kb, Vt);
  attention_kernel<<<dim3(NTOK / 128, BATCH * NH), blk, 0, stream>>>(
      Qb, Kb, Vt, xo);
  out_proj_kernel<<<dim3((BATCH * NTOK) / 128, DIM / 64), blk, 0, stream>>>(
      xo, Wo, bo, out);
}
